// FourierKANLayer_8100308320585
// MI455X (gfx1250) — matrix operand order repaired and checked by reading, not compiled, not run on hardware
//
#include <hip/hip_runtime.h>

#define LN_EPS 1e-5f
#define INV2PI 0.15915494309189535f

typedef __attribute__((ext_vector_type(16))) _Float16 v16h;
typedef __attribute__((ext_vector_type(8)))  _Float16 v8h;
typedef __attribute__((ext_vector_type(8)))  float    v8f;

#if defined(__has_builtin)
#if __has_builtin(__builtin_amdgcn_cosf) && __has_builtin(__builtin_amdgcn_sinf)
#define FKAN_FAST_TRIG 1
#endif
#endif

// cos/sin with argument in revolutions (hardware-native units for v_cos_f32/v_sin_f32)
static __device__ __forceinline__ float fcos_rev(float r) {
#ifdef FKAN_FAST_TRIG
  return __builtin_amdgcn_cosf(r);
#else
  return __cosf(r * 6.283185307179586f);
#endif
}
static __device__ __forceinline__ float fsin_rev(float r) {
#ifdef FKAN_FAST_TRIG
  return __builtin_amdgcn_sinf(r);
#else
  return __sinf(r * 6.283185307179586f);
#endif
}

// N=4096, D_IN=512, D_OUT=512, G=32.
// Source weights: Wsrc[T][o][i][g] f32 (T=0 cos, T=1 sin).
// Workspace f16 layout (interleaved so one pointer + imm offsets covers the loop):
//   W16[o][i][T][g] : element index ((o*512 + i)*2 + T)*32 + g

// ---------------- Stage 1: f32 -> f16 conversion + cos/sin interleave ----------------
__global__ __launch_bounds__(256) void fkan_convert_w(const float* __restrict__ src,
                                                      _Float16* __restrict__ dst) {
  const uint32_t gid   = blockIdx.x * 256u + threadIdx.x;
  const uint32_t idxin = gid * 8u;                    // 8 elements, within one g-block of 32
  const uint32_t T     = idxin >> 23;                 // 512*512*32 = 2^23 elements per T
  const uint32_t oi    = (idxin >> 5) & 0x3FFFFu;     // (o*512 + i), 2^18 groups per T
  const uint32_t g0    = idxin & 31u;                 // 0,8,16,24
  const uint32_t idxout = (oi << 6) + (T << 5) + g0;

  const float4* p = reinterpret_cast<const float4*>(src + idxin);
  float4 a = p[0], b = p[1];
  v8h r;
  r[0] = (_Float16)a.x; r[1] = (_Float16)a.y; r[2] = (_Float16)a.z; r[3] = (_Float16)a.w;
  r[4] = (_Float16)b.x; r[5] = (_Float16)b.y; r[6] = (_Float16)b.z; r[7] = (_Float16)b.w;
  *reinterpret_cast<v8h*>(dst + idxout) = r;
}

// ---------------- Stage 2: fused LayerNorm + Fourier-feature WMMA GEMM ----------------
// Block: 256 threads = 8 waves. Tile: 128 rows x 128 cols.
//   wave wv owns row tile wv*16 and all 128 block columns (8 WMMA col tiles)
//   => 16 WMMAs amortize each A-tile (cos/sin) construction.
// Grid: (512/128, 4096/128) = (4, 32).
template <bool PRE>
__global__ __launch_bounds__(256) void fkan_gemm(
    const float* __restrict__ x, const float* __restrict__ lnw, const float* __restrict__ lnb,
    const float* __restrict__ wf32, const _Float16* __restrict__ wf16,
    const float* __restrict__ bias, float* __restrict__ out) {
  __shared__ float ldsX[512 * 129];  // xn transposed [i][m], stride 129 (264 KB, bank-safe)

  const int tid  = threadIdx.x;
  const int lane = tid & 31;
  const int wv   = tid >> 5;
  const int by   = blockIdx.y;  // row block (128 rows)
  const int bx   = blockIdx.x;  // col block (128 cols)

  // ---- LayerNorm prologue: each wave normalizes 16 rows ----
  for (int rr = 0; rr < 16; ++rr) {
    const int m = wv * 16 + rr;  // 0..127 within block
    const float* xr = x + (size_t)(by * 128 + m) * 512u;
    float v[16];
    float s = 0.f, s2 = 0.f;
#pragma unroll
    for (int j = 0; j < 16; ++j) {
      v[j] = xr[lane + 32 * j];
      s += v[j];
      s2 += v[j] * v[j];
    }
#pragma unroll
    for (int off = 16; off >= 1; off >>= 1) {
      s  += __shfl_xor(s, off, 32);
      s2 += __shfl_xor(s2, off, 32);
    }
    const float mu   = s * (1.f / 512.f);
    const float rstd = rsqrtf(s2 * (1.f / 512.f) - mu * mu + LN_EPS);
#pragma unroll
    for (int j = 0; j < 16; ++j) {
      const int i = lane + 32 * j;
      ldsX[i * 129 + m] = (v[j] - mu) * rstd * lnw[i] + lnb[i];
    }
  }
  __syncthreads();

  const int h   = lane >> 4;   // lane half: K sub-range select in A/B layouts
  const int l16 = lane & 15;   // M row (A) / N column (B, D)
  const int xidx    = wv * 16 + l16;   // lds m-index for this lane's A row
  const int colbase = bx * 128;

  v8f acc[8];
#pragma unroll
  for (int t = 0; t < 8; ++t) acc[t] = v8f{0.f, 0.f, 0.f, 0.f, 0.f, 0.f, 0.f, 0.f};

  // PRE path: single per-lane pointer; every load is wp + compile-time element offset:
  //   tile t -> + t*524288 ; sin -> +32 ; per-i step -> +64
  const _Float16* wp =
      wf16 + (((uint32_t)(colbase + l16)) << 15) + (uint32_t)(h << 4);

  // non-PRE path offsets into original f32 layout: (col*512 + i)*32 + h*16 (+8M for sin)
  uint32_t wo[8];
#pragma unroll
  for (int t = 0; t < 8; ++t) {
    const uint32_t col = (uint32_t)(colbase + t * 16 + l16);
    wo[t] = (col << 14) + (uint32_t)(h << 4);
  }
  const uint32_t SIN_OFF = 512u << 14;

  const float k0f = (float)(16 * h + 1);  // first k in this lane's segment 1

  for (int i = 0; i < 512; ++i) {
    const float xv  = ldsX[i * 129 + xidx];
    const float rev = xv * INV2PI;

    // Chebyshev: f((k+1)x) = 2cos(x) f(kx) - f((k-1)x).  Two 8-long segments per lane:
    //   segment A: k = 16h+1 .. 16h+8   -> v16h elements 0..7
    //   segment B: k = 16h+9 .. 16h+16  -> v16h elements 8..15
    const float t2 = 2.f * fcos_rev(rev);
    const float a0 = k0f * rev, a1 = a0 + rev;
    const float b0 = a0 + 8.f * rev, b1 = b0 + rev;
    float ca[8], sa[8], cb[8], sb[8];
    ca[0] = fcos_rev(a0); ca[1] = fcos_rev(a1);
    sa[0] = fsin_rev(a0); sa[1] = fsin_rev(a1);
    cb[0] = fcos_rev(b0); cb[1] = fcos_rev(b1);
    sb[0] = fsin_rev(b0); sb[1] = fsin_rev(b1);
#pragma unroll
    for (int j = 2; j < 8; ++j) {
      ca[j] = t2 * ca[j - 1] - ca[j - 2];
      sa[j] = t2 * sa[j - 1] - sa[j - 2];
      cb[j] = t2 * cb[j - 1] - cb[j - 2];
      sb[j] = t2 * sb[j - 1] - sb[j - 2];
    }
    v16h ac, as;
#pragma unroll
    for (int e = 0; e < 8; ++e) {
      ac[e]     = (_Float16)ca[e];
      ac[e + 8] = (_Float16)cb[e];
      as[e]     = (_Float16)sa[e];
      as[e + 8] = (_Float16)sb[e];
    }

    if (PRE) {
#pragma unroll
      for (int t = 0; t < 8; ++t) {
        const v16h bc = *reinterpret_cast<const v16h*>(wp + t * 524288u);
        const v16h bs = *reinterpret_cast<const v16h*>(wp + t * 524288u + 32u);
        acc[t] = __builtin_amdgcn_wmma_f32_16x16x32_f16(false, ac, false, bc, (short)0,
                                                        acc[t], false, false);
        acc[t] = __builtin_amdgcn_wmma_f32_16x16x32_f16(false, as, false, bs, (short)0,
                                                        acc[t], false, false);
      }
      wp += 64;
    } else {
      const uint32_t istep = (uint32_t)(i * 32);
#pragma unroll
      for (int t = 0; t < 8; ++t) {
        v16h bc, bs;
        const float4* pc = reinterpret_cast<const float4*>(wf32 + (wo[t] + istep));
        const float4* ps = reinterpret_cast<const float4*>(wf32 + (wo[t] + SIN_OFF + istep));
#pragma unroll
        for (int q = 0; q < 4; ++q) {
          float4 fc = pc[q], fs = ps[q];
          bc[4 * q + 0] = (_Float16)fc.x; bc[4 * q + 1] = (_Float16)fc.y;
          bc[4 * q + 2] = (_Float16)fc.z; bc[4 * q + 3] = (_Float16)fc.w;
          bs[4 * q + 0] = (_Float16)fs.x; bs[4 * q + 1] = (_Float16)fs.y;
          bs[4 * q + 2] = (_Float16)fs.z; bs[4 * q + 3] = (_Float16)fs.w;
        }
        acc[t] = __builtin_amdgcn_wmma_f32_16x16x32_f16(false, ac, false, bc, (short)0,
                                                        acc[t], false, false);
        acc[t] = __builtin_amdgcn_wmma_f32_16x16x32_f16(false, as, false, bs, (short)0,
                                                        acc[t], false, false);
      }
    }
  }

  // ---- Epilogue: add bias, store per 16x16 f32 C/D layout (lane: M = 8h+v, N = l16) ----
#pragma unroll
  for (int t = 0; t < 8; ++t) {
    const int col = colbase + t * 16 + l16;
    const float bv = bias[col];
#pragma unroll
    for (int vv = 0; vv < 8; ++vv) {
      const int row = by * 128 + wv * 16 + h * 8 + vv;
      out[(size_t)row * 512u + col] = acc[t][vv] + bv;
    }
  }
}

extern "C" void kernel_launch(void* const* d_in, const int* in_sizes, int n_in,
                              void* d_out, int out_size, void* d_ws, size_t ws_size,
                              hipStream_t stream) {
  const float* x    = (const float*)d_in[0];
  const float* lnw  = (const float*)d_in[1];
  const float* lnb  = (const float*)d_in[2];
  const float* wf   = (const float*)d_in[3];  // [2,512,512,32] f32
  const float* bias = (const float*)d_in[4];
  float* out = (float*)d_out;

  const size_t WN = (size_t)2 * 512 * 512 * 32;  // 16,777,216 weight elements
  dim3 grid(4, 32), block(256);

  if (ws_size >= WN * sizeof(_Float16)) {
    _Float16* w16 = (_Float16*)d_ws;
    fkan_convert_w<<<(unsigned)(WN / (256 * 8)), 256, 0, stream>>>(wf, w16);
    fkan_gemm<true><<<grid, block, 0, stream>>>(x, lnw, lnb, wf, w16, bias, out);
  } else {
    fkan_gemm<false><<<grid, block, 0, stream>>>(x, lnw, lnb, wf, (const _Float16*)nullptr,
                                                 bias, out);
  }
}
